// EGNN_77360950935889
// MI455X (gfx1250) — matrix operand order repaired and checked by reading, not compiled, not run on hardware
//
#include <hip/hip_runtime.h>
#include <hip/hip_bf16.h>
#include <stdint.h>

typedef __attribute__((ext_vector_type(16))) _Float16 v16h;
typedef __attribute__((ext_vector_type(8)))  float    v8f;

// ---------------------------------------------------------------------------
// Utility kernels
// ---------------------------------------------------------------------------
__global__ void gcn_zero_f32(float* __restrict__ p, long long n) {
    long long i = (long long)blockIdx.x * blockDim.x + threadIdx.x;
    if (i < n) p[i] = 0.0f;
}

__global__ void gcn_deg_count(const int* __restrict__ dst, int E,
                              float* __restrict__ degf) {
    long long i = (long long)blockIdx.x * blockDim.x + threadIdx.x;
    if (i < E) atomicAdd(&degf[dst[i]], 1.0f);
}

__global__ void gcn_dinv(const float* __restrict__ degf,
                         float* __restrict__ dinv, int N) {
    int i = blockIdx.x * blockDim.x + threadIdx.x;
    if (i < N) {
        float d = degf[i] + 1.0f;   // +1 self-loop; always > 0
        dinv[i] = rsqrtf(d);
    }
}

// ---------------------------------------------------------------------------
// Dense transform Ht[N,Fout] = H[N,Fin] @ W[Fin,Fout] via v_wmma_f32_16x16x32_f16
// Covers whole 16-row tiles only (branch-free loads AND stores).
// Block = 256 threads = 8 waves; each wave owns one 16-row node tile.
// B fragments pre-swizzled into LDS -> one aligned v16h (2x ds_load_b128) each.
// ---------------------------------------------------------------------------
template <int Fin, int Fout>
__global__ void gcn_transform(const float* __restrict__ H,
                              const float* __restrict__ W,
                              float* __restrict__ Ht, int Ntiles) {
    constexpr int KC = (Fin + 31) / 32;   // K chunks of 32
    constexpr int NT = Fout / 16;         // 16-column output tiles

    // B fragments in exact WMMA lane layout: [nt][kc][lane][elem0..15]
    __shared__ __align__(32) _Float16 sB[NT * KC * 32 * 16];
    for (int i = threadIdx.x; i < NT * KC * 32 * 16; i += blockDim.x) {
        int t    = i;
        int elem = t & 15; t >>= 4;
        int ln   = t & 31; t >>= 5;
        int kc   = t % KC;
        int nt   = t / KC;
        int hi   = ln >> 4;
        int v    = elem >> 1;
        int kb   = kc * 32 + ((v >= 4) ? 16 : 0) + hi * 8 + (elem & 7);
        int ncol = nt * 16 + (ln & 15);
        sB[i] = (kb < Fin) ? (_Float16)W[kb * Fout + ncol] : (_Float16)0.0f;
    }
    __syncthreads();

    const int wave = threadIdx.x >> 5;
    const int lane = threadIdx.x & 31;
    const int m    = lane & 15;     // A-matrix row within tile (both lane halves)
    const int hi   = lane >> 4;     // lane half selects K sub-range
    const int tile = blockIdx.x * (blockDim.x >> 5) + wave;
    if (tile >= Ntiles) return;     // wave-uniform exit: EXEC stays all-ones
    const int row0 = tile * 16;

    const float* __restrict__ rp = H + (size_t)(row0 + m) * Fin;

    // A fragments: VGPR v holds K = kc*32 + (v>=4?16:0) + (v&3)*2 + hi*8, +1
    v16h a[KC];
#pragma unroll
    for (int kc = 0; kc < KC; ++kc) {
#pragma unroll
        for (int v = 0; v < 8; ++v) {
            const int kb = kc * 32 + ((v >= 4) ? 16 : 0) + (v & 3) * 2 + hi * 8;
            if (kb + 1 < Fin) {              // per-lane: kb depends on lane half hi
                const float2 fv = *(const float2*)(rp + kb);
                a[kc][2 * v]     = (_Float16)fv.x;
                a[kc][2 * v + 1] = (_Float16)fv.y;
            } else if (kb < Fin) {
                a[kc][2 * v]     = (_Float16)rp[kb];
                a[kc][2 * v + 1] = (_Float16)0.0f;
            } else {
                a[kc][2 * v]     = (_Float16)0.0f;
                a[kc][2 * v + 1] = (_Float16)0.0f;
            }
        }
    }

    // Per-lane store base: row (row0 + hi*8), column (nt*16 + lane&15).
    float* __restrict__ outBase = Ht + (size_t)(row0 + hi * 8) * Fout + (lane & 15);

#pragma unroll
    for (int nt = 0; nt < NT; ++nt) {
        v8f c = {};
#pragma unroll
        for (int kc = 0; kc < KC; ++kc) {
            const v16h b = *(const v16h*)&sB[(((nt * KC) + kc) * 32 + lane) * 16];
            c = __builtin_amdgcn_wmma_f32_16x16x32_f16(
                    false, a[kc], false, b, (short)0, c, false, false);
        }
        // D layout: VGPR r -> M = r + hi*8, N = lane&15.
        // Unconditional stores: one address, immediate offsets r*Fout*4.
#pragma unroll
        for (int r = 0; r < 8; ++r)
            outBase[(size_t)r * Fout + nt * 16] = c[r];
    }
}

// Tail rows [n0, N): trivial scalar matmul (never launched when N % 16 == 0).
template <int Fin, int Fout>
__global__ void gcn_transform_tail(const float* __restrict__ H,
                                   const float* __restrict__ W,
                                   float* __restrict__ Ht, int n0, int N) {
    int idx = blockIdx.x * blockDim.x + threadIdx.x;
    const int total = (N - n0) * Fout;
    if (idx >= total) return;
    const int r = n0 + idx / Fout;
    const int c = idx % Fout;
    float acc = 0.0f;
#pragma unroll
    for (int k = 0; k < Fin; ++k)
        acc += (float)(_Float16)H[(size_t)r * Fin + k] * (float)(_Float16)W[k * Fout + c];
    Ht[(size_t)r * Fout + c] = acc;
}

// ---------------------------------------------------------------------------
// acc[i,f] = Ht[i,f] * dinv[i]^2 + b[f]      (self-loop term + bias)
// ---------------------------------------------------------------------------
template <int F>
__global__ void gcn_init(const float* __restrict__ Ht,
                         const float* __restrict__ bias,
                         const float* __restrict__ dinv,
                         float* __restrict__ acc, int N) {
    long long idx = (long long)blockIdx.x * blockDim.x + threadIdx.x;
    if (idx >= (long long)N * F) return;
    const int i = (int)(idx / F);
    const int f = (int)(idx % F);
    const float di = dinv[i];
    acc[idx] = Ht[idx] * di * di + bias[f];
}

// ---------------------------------------------------------------------------
// Edge scatter: F/4 lanes per edge, float4 gather, f32 atomic scatter (L2 RMW)
// ---------------------------------------------------------------------------
template <int F>
__global__ void gcn_scatter(const int* __restrict__ src,
                            const int* __restrict__ dst,
                            const float* __restrict__ Ht,
                            const float* __restrict__ dinv,
                            float* __restrict__ acc, int E) {
    constexpr int TPE = F / 4;   // power of two -> shifts
    long long tid = (long long)blockIdx.x * blockDim.x + threadIdx.x;
    const int e = (int)(tid / TPE);
    const int c = (int)(tid % TPE);
    if (e >= E) return;
    __builtin_prefetch(&src[e + 2048], 0, 0);   // global_prefetch_b8 (speculative)
    __builtin_prefetch(&dst[e + 2048], 0, 0);
    const int s = src[e];
    const int d = dst[e];
    const float w = dinv[s] * dinv[d];
    const float4 v = *(const float4*)(Ht + (size_t)s * F + c * 4);
    float* out = acc + (size_t)d * F + c * 4;
    atomicAdd(out + 0, v.x * w);
    atomicAdd(out + 1, v.y * w);
    atomicAdd(out + 2, v.z * w);
    atomicAdd(out + 3, v.w * w);
}

__global__ void gcn_relu(float* __restrict__ p, long long n) {
    long long i = (long long)blockIdx.x * blockDim.x + threadIdx.x;
    if (i < n) p[i] = fmaxf(p[i], 0.0f);
}

// ---------------------------------------------------------------------------
// global_mean_pool accumulation + final linear layer
// ---------------------------------------------------------------------------
__global__ void gcn_pool(const float* __restrict__ h,
                         const int* __restrict__ batch,
                         float* __restrict__ sums,
                         float* __restrict__ cnts, int N) {
    long long idx = (long long)blockIdx.x * blockDim.x + threadIdx.x;
    if (idx >= (long long)N * 16) return;
    const int i = (int)(idx >> 4);
    const int f = (int)(idx & 15);
    const int g = batch[i];
    atomicAdd(&sums[g * 16 + f], h[idx]);
    if (f == 0) atomicAdd(&cnts[g], 1.0f);
}

__global__ void gcn_final(const float* __restrict__ sums,
                          const float* __restrict__ cnts,
                          const float* __restrict__ Wl,
                          const float* __restrict__ bl,
                          float* __restrict__ out, int G) {
    int g = blockIdx.x * blockDim.x + threadIdx.x;
    if (g >= G) return;
    const float inv = 1.0f / fmaxf(cnts[g], 1.0f);
    float accv = bl[0];
#pragma unroll
    for (int f = 0; f < 16; ++f)
        accv += (sums[g * 16 + f] * inv) * Wl[f];
    out[g] = fmaxf(accv, 0.0f);
}

// ---------------------------------------------------------------------------
// Launcher
// ---------------------------------------------------------------------------
static inline int cdiv_ll(long long a, int b) { return (int)((a + b - 1) / b); }

extern "C" void kernel_launch(void* const* d_in, const int* in_sizes, int n_in,
                              void* d_out, int out_size, void* d_ws, size_t ws_size,
                              hipStream_t stream) {
    const float* x    = (const float*)d_in[0];
    const float* W1   = (const float*)d_in[1];
    const float* b1   = (const float*)d_in[2];
    const float* W2   = (const float*)d_in[3];
    const float* b2   = (const float*)d_in[4];
    const float* W3   = (const float*)d_in[5];
    const float* b3   = (const float*)d_in[6];
    const float* Wl   = (const float*)d_in[7];
    const float* bl   = (const float*)d_in[8];
    const int*   eidx = (const int*)d_in[9];
    const int*   batch= (const int*)d_in[10];
    float*       out  = (float*)d_out;

    const int N = in_sizes[0] / 4;       // 100000
    const int E = in_sizes[9] / 2;       // 3200000
    const int G = out_size;              // 1000
    const int* src = eidx;
    const int* dst = eidx + E;

    const int Ntiles = N / 16;           // whole tiles (6250)
    const int N16    = Ntiles * 16;
    const int rem    = N - N16;          // 0 for N = 100000

    float* wsf  = (float*)d_ws;
    float* degf = wsf;                         // N
    float* dinv = degf + N;                    // N
    float* bufA = dinv + N;                    // N*64 (transform out)
    float* bufB = bufA + (size_t)N * 64;       // N*64
    float* bufC = bufB + (size_t)N * 64;       // N*64
    float* sums = bufC + (size_t)N * 64;       // G*16
    float* cnts = sums + (size_t)G * 16;       // G   (contiguous after sums)

    const int B256 = 256;
    const int TB   = 256;                      // transform block: 8 waves
    const int wavesPerBlock = TB / 32;
    const int tgrid = (Ntiles + wavesPerBlock - 1) / wavesPerBlock;

    // degrees -> dinv
    gcn_zero_f32<<<cdiv_ll(N, B256), B256, 0, stream>>>(degf, N);
    gcn_deg_count<<<cdiv_ll(E, B256), B256, 0, stream>>>(dst, E, degf);
    gcn_dinv<<<cdiv_ll(N, B256), B256, 0, stream>>>(degf, dinv, N);

    // ---- Layer 1: 4 -> 64, ReLU ----
    gcn_transform<4, 64><<<tgrid, TB, 0, stream>>>(x, W1, bufA, Ntiles);
    if (rem) gcn_transform_tail<4, 64><<<cdiv_ll((long long)rem * 64, B256), B256, 0, stream>>>(x, W1, bufA, N16, N);
    gcn_init<64><<<cdiv_ll((long long)N * 64, B256), B256, 0, stream>>>(bufA, b1, dinv, bufB, N);
    gcn_scatter<64><<<cdiv_ll((long long)E * 16, B256), B256, 0, stream>>>(src, dst, bufA, dinv, bufB, E);
    gcn_relu<<<cdiv_ll((long long)N * 64, B256), B256, 0, stream>>>(bufB, (long long)N * 64);

    // ---- Layer 2: 64 -> 32, ReLU ----
    gcn_transform<64, 32><<<tgrid, TB, 0, stream>>>(bufB, W2, bufA, Ntiles);
    if (rem) gcn_transform_tail<64, 32><<<cdiv_ll((long long)rem * 32, B256), B256, 0, stream>>>(bufB, W2, bufA, N16, N);
    gcn_init<32><<<cdiv_ll((long long)N * 32, B256), B256, 0, stream>>>(bufA, b2, dinv, bufC, N);
    gcn_scatter<32><<<cdiv_ll((long long)E * 8, B256), B256, 0, stream>>>(src, dst, bufA, dinv, bufC, E);
    gcn_relu<<<cdiv_ll((long long)N * 32, B256), B256, 0, stream>>>(bufC, (long long)N * 32);

    // ---- Layer 3: 32 -> 16, no ReLU ----
    gcn_transform<32, 16><<<tgrid, TB, 0, stream>>>(bufC, W3, bufA, Ntiles);
    if (rem) gcn_transform_tail<32, 16><<<cdiv_ll((long long)rem * 16, B256), B256, 0, stream>>>(bufC, W3, bufA, N16, N);
    gcn_init<16><<<cdiv_ll((long long)N * 16, B256), B256, 0, stream>>>(bufA, b3, dinv, bufB, N);
    gcn_scatter<16><<<cdiv_ll((long long)E * 4, B256), B256, 0, stream>>>(src, dst, bufA, dinv, bufB, E);

    // ---- mean pool + final linear ----
    gcn_zero_f32<<<cdiv_ll((long long)G * 17, B256), B256, 0, stream>>>(sums, (long long)G * 17);
    gcn_pool<<<cdiv_ll((long long)N * 16, B256), B256, 0, stream>>>(bufB, batch, sums, cnts, N);
    gcn_final<<<cdiv_ll(G, B256), B256, 0, stream>>>(sums, cnts, Wl, bl, out, G);
}
